// HierarchicalMixtureOfExperts_82231443849803
// MI455X (gfx1250) — matrix operand order repaired and checked by reading, not compiled, not run on hardware
//
#include <hip/hip_runtime.h>
#include <hip/hip_bf16.h>
#include <math.h>

// ---------------------------------------------------------------------------
// HierarchicalMixtureOfExperts forward for MI455X (gfx1250, wave32, WMMA).
// GEMMs use v_wmma_f32_16x16x32_f16; fp32 weights convert to f16 while staging
// to LDS (weight streaming ~288MB fp32 -> ~12us HBM floor at 23.3 TB/s is the
// bound; WMMA compute sits far under it). A-operand tiles (pure f16, no
// conversion) are staged by the Tensor Data Mover (tensor_load_to_lds) with
// TDM padding reproducing the LDS row pitch; completion via s_wait_tensorcnt.
// ---------------------------------------------------------------------------

typedef __attribute__((ext_vector_type(16))) _Float16 v16h;
typedef __attribute__((ext_vector_type(8)))  _Float16 v8h;
typedef __attribute__((ext_vector_type(8)))  float    v8f;
typedef __attribute__((ext_vector_type(4)))  unsigned int u32x4;
typedef __attribute__((ext_vector_type(8)))  int      i32x8;
typedef __attribute__((ext_vector_type(4)))  int      i32x4;

#ifndef __has_builtin
#define __has_builtin(x) 0
#endif
#if defined(__AMDGCN__) && __has_builtin(__builtin_amdgcn_tensor_load_to_lds) && \
    __has_builtin(__builtin_amdgcn_s_wait_tensorcnt)
#define HAVE_TDM 1
#else
#define HAVE_TDM 0
#endif

#define LN_EPS 1e-5f

__device__ __forceinline__ float gelu_exact(float x) {
    return 0.5f * x * (1.0f + erff(x * 0.70710678118654752440f));
}

// ---------------------------------------------------------------------------
// prep: h = x + pos_encoding; shared LN stats of h -> xhat (f16);
// router LN -> rnorm = xhat*r_g + r_b (f16). One block per token, D = 1024.
// ---------------------------------------------------------------------------
__global__ __launch_bounds__(256)
void prep_kernel(const float* __restrict__ x,
                 const float* __restrict__ r_g, const float* __restrict__ r_b,
                 float* __restrict__ h,
                 _Float16* __restrict__ xhat,
                 _Float16* __restrict__ rnorm) {
    const int D = 1024, S = 1024;
    const int token = blockIdx.x;          // b*S + s
    const int s = token % S;
    const int t = threadIdx.x;             // 256 threads, 4 elements each
    __shared__ float red[256];

    float v[4];
    float sum = 0.f;
#pragma unroll
    for (int i = 0; i < 4; ++i) {
        int d = t + i * 256;
        float fd  = (float)(d & ~1);
        float div = expf(-logf(10000.0f) * fd / (float)D);
        float ang = (float)s * div;
        float pe  = (d & 1) ? cosf(ang) : sinf(ang);
        float val = x[(size_t)token * D + d] + pe;
        v[i] = val;
        sum += val;
    }
    red[t] = sum; __syncthreads();
#pragma unroll
    for (int off = 128; off > 0; off >>= 1) {
        if (t < off) red[t] += red[t + off];
        __syncthreads();
    }
    float mean = red[0] * (1.0f / (float)D);
    __syncthreads();

    float vs = 0.f;
#pragma unroll
    for (int i = 0; i < 4; ++i) { float d0 = v[i] - mean; vs += d0 * d0; }
    red[t] = vs; __syncthreads();
#pragma unroll
    for (int off = 128; off > 0; off >>= 1) {
        if (t < off) red[t] += red[t + off];
        __syncthreads();
    }
    float rstd = rsqrtf(red[0] * (1.0f / (float)D) + LN_EPS);
    __syncthreads();

#pragma unroll
    for (int i = 0; i < 4; ++i) {
        int d = t + i * 256;
        size_t idx = (size_t)token * D + d;
        float xh = (v[i] - mean) * rstd;
        h[idx]     = v[i];
        xhat[idx]  = (_Float16)xh;
        rnorm[idx] = (_Float16)(xh * r_g[d] + r_b[d]);
    }
}

// ---------------------------------------------------------------------------
// Generic WMMA GEMM: C[M,N] = epi(A[M,K](f16) * Bw[K,N](f32->f16) + bias)
//   EPI 1: GELU -> f16;  EPI 2: plain -> f32;
//   EPI 3: Cout (+)= gate[row*gstride] * (acc + bias)  (f32)
//   TDMA:  stage A tile with the Tensor Data Mover (requires ag == nullptr)
// Block tile 128x128, K-step 32. 8 waves; wave tile 64x32 (4x2 WMMA tiles).
// ---------------------------------------------------------------------------
template <int EPI, bool INIT, bool TDMA>
__global__ __launch_bounds__(256)
void gemm_wmma(const _Float16* __restrict__ A, int K,
               const float* __restrict__ Bw, int N,
               const float* __restrict__ bias,
               const float* __restrict__ ag, const float* __restrict__ ab,
               const float* __restrict__ gate, int gstride,
               void* __restrict__ Cout) {
    const int BM = 128, BN = 128, BK = 32;
    __shared__ _Float16 sA[128 * 40];    // [row][k], row pitch 40 f16 (80B)
    __shared__ _Float16 sB[32 * 136];    // [k][n],  row pitch 136 f16 (272B)

    const int tid  = threadIdx.x;
    const int lane = tid & 31;
    const int wave = tid >> 5;           // 0..7
    const int wm   = wave >> 2;          // 0..1 : 64-row slab
    const int wn   = wave & 3;           // 0..3 : 32-col slab
    const int lo   = lane & 15;
    const int hi   = lane >> 4;

    const int row0 = blockIdx.y * BM;
    const int col0 = blockIdx.x * BN;
    const unsigned Mdim = gridDim.y * BM;

    // manual global->LDS coords
    const int a_row  = tid >> 1;         // 0..127
    const int a_half = tid & 1;          // two 16-f16 halves per A row
    const int b_row  = tid >> 3;         // 0..31
    const int b_col  = (tid & 7) * 16;   // 0..112

    v8f acc[4][2];
#pragma unroll
    for (int tm = 0; tm < 4; ++tm)
#pragma unroll
        for (int tn = 0; tn < 2; ++tn)
#pragma unroll
            for (int r = 0; r < 8; ++r) acc[tm][tn][r] = 0.0f;

    for (int k0 = 0; k0 < K; k0 += BK) {
        // ---- stage A tile ----
        if (TDMA) {
#if HAVE_TDM
            if (wave == 0) {
                // Tensor DMA: 2D f16 tile [128 rows x 32 cols] of A[M,K], with
                // LDS padding 4 DWORDs every 16 DWORDs -> row pitch 40 f16.
                unsigned long long ga = (unsigned long long)(uintptr_t)A +
                    2ull * ((unsigned long long)row0 * (unsigned)K + (unsigned)k0);
                unsigned lds = (unsigned)(uintptr_t)(void*)sA;  // group-segment offset
                u32x4 g0;
                g0.x = 1u;                                   // count=1 (valid user D#)
                g0.y = lds;                                  // lds_addr
                g0.z = (unsigned)(ga & 0xFFFFFFFFu);         // global_addr[31:0]
                g0.w = (unsigned)((ga >> 32) & 0x01FFFFFFu) | (2u << 30);  // [56:32] | type=2
                i32x8 g1;
                g1[0] = (1 << 16)            // data_size = 2 bytes
                      | (1 << 20)            // pad_enable
                      | (3 << 22)            // pad_interval: 16 DWORDs (64B = one row)
                      | (3 << 25);           // pad_amount: 4 DWORDs (16B)
                g1[1] = (int)(((unsigned)K & 0xFFFFu) << 16);            // tensor_dim0[15:0]
                g1[2] = (int)(((unsigned)K >> 16) | ((Mdim & 0xFFFFu) << 16)); // dim0 hi | dim1 lo
                g1[3] = (int)((Mdim >> 16) | (32u << 16));               // dim1 hi | tile_dim0=32
                g1[4] = 128;                                             // tile_dim1=128, tile_dim2=0
                g1[5] = K;                                               // tensor_dim0_stride[31:0]
                g1[6] = 0;                                               // stride hi | dim1_stride lo
                g1[7] = 0;
                i32x4 gz4 = {0, 0, 0, 0};                // 2D: groups 2/3 unused
                i32x8 gz8 = {0, 0, 0, 0, 0, 0, 0, 0};    // trailing group unused
                __builtin_amdgcn_tensor_load_to_lds(g0, g1, gz4, gz4, gz8, 0);
            }
#else
            {   // fallback: manual copy (no affine on this path)
                const _Float16* src = A + (size_t)(row0 + a_row) * K + k0 + a_half * 16;
                v8h p0 = *(const v8h*)(src);
                v8h p1 = *(const v8h*)(src + 8);
                *(v8h*)(&sA[a_row * 40 + a_half * 16])     = p0;
                *(v8h*)(&sA[a_row * 40 + a_half * 16 + 8]) = p1;
            }
#endif
        } else {
            // manual staging with per-K affine (per-expert LN fold-in)
            const _Float16* src = A + (size_t)(row0 + a_row) * K + k0 + a_half * 16;
            v8h p0 = *(const v8h*)(src);
            v8h p1 = *(const v8h*)(src + 8);
            if (ag) {
                int kb = k0 + a_half * 16;
#pragma unroll
                for (int i = 0; i < 8; ++i) {
                    p0[i] = (_Float16)((float)p0[i] * ag[kb + i]     + ab[kb + i]);
                    p1[i] = (_Float16)((float)p1[i] * ag[kb + 8 + i] + ab[kb + 8 + i]);
                }
            }
            *(v8h*)(&sA[a_row * 40 + a_half * 16])     = p0;
            *(v8h*)(&sA[a_row * 40 + a_half * 16 + 8]) = p1;
        }

        // ---- stage B tile (fp32 -> f16, K-major, all 256 threads) ----
        {
            const float* src = Bw + (size_t)(k0 + b_row) * N + col0 + b_col;
            if (k0 + BK < K)
                __builtin_prefetch(src + (size_t)BK * N, 0, 1);  // global_prefetch_b8
            v8h q0, q1;
#pragma unroll
            for (int i = 0; i < 8; ++i) {
                q0[i] = (_Float16)src[i];
                q1[i] = (_Float16)src[8 + i];
            }
            *(v8h*)(&sB[b_row * 136 + b_col])     = q0;
            *(v8h*)(&sB[b_row * 136 + b_col + 8]) = q1;
        }

#if HAVE_TDM
        if (TDMA && wave == 0) __builtin_amdgcn_s_wait_tensorcnt(0);
#endif
        __syncthreads();

        // ---- fragments (ISA 16-bit A/B VGPR layouts) ----
        v16h af[4], bf[2];
#pragma unroll
        for (int tm = 0; tm < 4; ++tm) {
            int r = wm * 64 + tm * 16 + lo;           // A: lane lo = M row
            v8h l8 = *(const v8h*)&sA[r * 40 + hi * 8];        // K = hi*8 + 0..7
            v8h h8 = *(const v8h*)&sA[r * 40 + hi * 8 + 16];   // K = hi*8 + 16..23
#pragma unroll
            for (int i = 0; i < 8; ++i) { af[tm][i] = l8[i]; af[tm][8 + i] = h8[i]; }
        }
#pragma unroll
        for (int tn = 0; tn < 2; ++tn) {
            int c = wn * 32 + tn * 16;                // B: lane lo = N col, K = hi*8 + 0..7, hi*8 + 16..23
#pragma unroll
            for (int i = 0; i < 8; ++i) { bf[tn][i] = sB[(hi * 8 + i) * 136 + c + lo]; bf[tn][8 + i] = sB[(hi * 8 + 16 + i) * 136 + c + lo]; }
        }

        // ---- 8 WMMAs per K-step ----
#pragma unroll
        for (int tm = 0; tm < 4; ++tm)
#pragma unroll
            for (int tn = 0; tn < 2; ++tn)
                acc[tm][tn] = __builtin_amdgcn_wmma_f32_16x16x32_f16(
                    false, af[tm], false, bf[tn],
                    (short)0, acc[tm][tn], false, false);
        __syncthreads();
    }

    // ---- epilogue ----
#pragma unroll
    for (int tm = 0; tm < 4; ++tm) {
#pragma unroll
        for (int tn = 0; tn < 2; ++tn) {
            int colb = col0 + wn * 32 + tn * 16 + lo;   // C/D: lane lo = N col
            float bv = bias ? bias[colb] : 0.0f;
#pragma unroll
            for (int r = 0; r < 8; ++r) {               // VGPR r -> M = hi*8 + r
                int row = row0 + wm * 64 + tm * 16 + hi * 8 + r;
                float val = acc[tm][tn][r] + bv;
                if (EPI == 1) {
                    ((_Float16*)Cout)[(size_t)row * N + colb] = (_Float16)gelu_exact(val);
                } else if (EPI == 2) {
                    ((float*)Cout)[(size_t)row * N + colb] = val;
                } else {  // EPI == 3 : gated accumulate
                    float g = gate[(size_t)row * gstride];
                    float* o = (float*)Cout + (size_t)row * N + colb;
                    float prev = INIT ? 0.0f : *o;
                    *o = prev + g * val;
                }
            }
        }
    }
}

// ---------------------------------------------------------------------------
// Router head: logits = ract[token,:] @ r_w2 + r_b2 (N=8, too small for WMMA),
// temperature softmax, top-2 sparse gate. One block per token.
// ---------------------------------------------------------------------------
__global__ __launch_bounds__(256)
void router_gate_kernel(const _Float16* __restrict__ ract,  // [M,4096]
                        const float* __restrict__ w2,       // [4096,8]
                        const float* __restrict__ b2,       // [8]
                        const float* __restrict__ temp,     // [1]
                        float* __restrict__ gate) {         // [M,8]
    const int H = 4096, E = 8;
    const int token = blockIdx.x;
    const int t = threadIdx.x;
    __shared__ float red[256];
    __shared__ float logits[8];

    float s[8];
#pragma unroll
    for (int e = 0; e < E; ++e) s[e] = 0.0f;
    for (int k = t; k < H; k += 256) {
        float a = (float)ract[(size_t)token * H + k];
        const float* wr = w2 + (size_t)k * E;
#pragma unroll
        for (int e = 0; e < E; ++e) s[e] += a * wr[e];
    }
#pragma unroll
    for (int e = 0; e < E; ++e) {
        red[t] = s[e]; __syncthreads();
#pragma unroll
        for (int off = 128; off > 0; off >>= 1) {
            if (t < off) red[t] += red[t + off];
            __syncthreads();
        }
        if (t == 0) logits[e] = (red[0] + b2[e]) / temp[0];
        __syncthreads();
    }
    if (t == 0) {
        float mx = logits[0];
#pragma unroll
        for (int e = 1; e < E; ++e) mx = fmaxf(mx, logits[e]);
        float ex[8], sum = 0.0f;
#pragma unroll
        for (int e = 0; e < E; ++e) { ex[e] = expf(logits[e] - mx); sum += ex[e]; }
        int i0 = 0;
#pragma unroll
        for (int e = 1; e < E; ++e) if (ex[e] > ex[i0]) i0 = e;
        int i1 = (i0 == 0) ? 1 : 0;
#pragma unroll
        for (int e = 0; e < E; ++e) if (e != i0 && ex[e] > ex[i1]) i1 = e;
        float inv = 1.0f / sum;
#pragma unroll
        for (int e = 0; e < E; ++e)
            gate[(size_t)token * E + e] = (e == i0 || e == i1) ? ex[e] * inv : 0.0f;
    }
}

// ---------------------------------------------------------------------------
// LayerNorm (f32 in) -> affine -> f16 out. One block per token, D = 1024.
// ---------------------------------------------------------------------------
__global__ __launch_bounds__(256)
void ln_f16_kernel(const float* __restrict__ in,
                   const float* __restrict__ g, const float* __restrict__ b,
                   _Float16* __restrict__ out) {
    const int D = 1024;
    const int token = blockIdx.x;
    const int t = threadIdx.x;
    __shared__ float red[256];
    float v[4], sum = 0.f;
#pragma unroll
    for (int i = 0; i < 4; ++i) { v[i] = in[(size_t)token * D + t + i * 256]; sum += v[i]; }
    red[t] = sum; __syncthreads();
#pragma unroll
    for (int off = 128; off > 0; off >>= 1) { if (t < off) red[t] += red[t + off]; __syncthreads(); }
    float mean = red[0] * (1.0f / (float)D);
    __syncthreads();
    float vs = 0.f;
#pragma unroll
    for (int i = 0; i < 4; ++i) { float d0 = v[i] - mean; vs += d0 * d0; }
    red[t] = vs; __syncthreads();
#pragma unroll
    for (int off = 128; off > 0; off >>= 1) { if (t < off) red[t] += red[t + off]; __syncthreads(); }
    float rstd = rsqrtf(red[0] * (1.0f / (float)D) + LN_EPS);
    __syncthreads();
#pragma unroll
    for (int i = 0; i < 4; ++i) {
        int d = t + i * 256;
        out[(size_t)token * D + d] = (_Float16)((v[i] - mean) * rstd * g[d] + b[d]);
    }
}

// ---------------------------------------------------------------------------
// final: out = LayerNorm(h + c, o_g, o_b)  (f32 out). One block per token.
// ---------------------------------------------------------------------------
__global__ __launch_bounds__(256)
void final_kernel(const float* __restrict__ h, const float* __restrict__ c,
                  const float* __restrict__ g, const float* __restrict__ b,
                  float* __restrict__ out) {
    const int D = 1024;
    const int token = blockIdx.x;
    const int t = threadIdx.x;
    __shared__ float red[256];
    float v[4], sum = 0.f;
#pragma unroll
    for (int i = 0; i < 4; ++i) {
        size_t idx = (size_t)token * D + t + i * 256;
        v[i] = h[idx] + c[idx]; sum += v[i];
    }
    red[t] = sum; __syncthreads();
#pragma unroll
    for (int off = 128; off > 0; off >>= 1) { if (t < off) red[t] += red[t + off]; __syncthreads(); }
    float mean = red[0] * (1.0f / (float)D);
    __syncthreads();
    float vs = 0.f;
#pragma unroll
    for (int i = 0; i < 4; ++i) { float d0 = v[i] - mean; vs += d0 * d0; }
    red[t] = vs; __syncthreads();
#pragma unroll
    for (int off = 128; off > 0; off >>= 1) { if (t < off) red[t] += red[t + off]; __syncthreads(); }
    float rstd = rsqrtf(red[0] * (1.0f / (float)D) + LN_EPS);
    __syncthreads();
#pragma unroll
    for (int i = 0; i < 4; ++i) {
        int d = t + i * 256;
        out[(size_t)token * D + d] = (v[i] - mean) * rstd * g[d] + b[d];
    }
}

// ---------------------------------------------------------------------------
extern "C" void kernel_launch(void* const* d_in, const int* in_sizes, int n_in,
                              void* d_out, int out_size, void* d_ws, size_t ws_size,
                              hipStream_t stream) {
    (void)in_sizes; (void)n_in; (void)out_size; (void)ws_size;
    const int Bb = 2, S = 1024, D = 1024, H = 4096, E = 8, HE = 4096;
    const int M = Bb * S;  // 2048 tokens

    const float* x      = (const float*)d_in[0];
    const float* r_ln_g = (const float*)d_in[1];
    const float* r_ln_b = (const float*)d_in[2];
    const float* r_w1   = (const float*)d_in[3];
    const float* r_b1   = (const float*)d_in[4];
    const float* r_w2   = (const float*)d_in[5];
    const float* r_b2   = (const float*)d_in[6];
    const float* temp   = (const float*)d_in[7];
    const float* e_ln_g = (const float*)d_in[8];
    const float* e_ln_b = (const float*)d_in[9];
    const float* e_w1   = (const float*)d_in[10];
    const float* e_b1   = (const float*)d_in[11];
    const float* e_w2   = (const float*)d_in[12];
    const float* e_b2   = (const float*)d_in[13];
    const float* c_ln_g = (const float*)d_in[14];
    const float* c_ln_b = (const float*)d_in[15];
    const float* c_w1   = (const float*)d_in[16];
    const float* c_b1   = (const float*)d_in[17];
    const float* c_w2   = (const float*)d_in[18];
    const float* c_b2   = (const float*)d_in[19];
    const float* o_ln_g = (const float*)d_in[20];
    const float* o_ln_b = (const float*)d_in[21];

    // workspace layout (bytes)
    char* ws = (char*)d_ws;
    float*    h     = (float*)   (ws + 0);                   //  8 MB [M,D] f32
    _Float16* xhat  = (_Float16*)(ws + 8388608);             //  4 MB [M,D] f16
    _Float16* rnorm = (_Float16*)(ws + 12582912);            //  4 MB [M,D] f16
    _Float16* ract  = (_Float16*)(ws + 16777216);            // 16 MB [M,H] f16
    float*    gate  = (float*)   (ws + 33554432);            // 64 KB [M,E] f32
    _Float16* f1    = (_Float16*)(ws + 33619968);            // 16 MB [M,HE] f16 (per-expert, reused)
    float*    comb  = (float*)   (ws + 50397184);            //  8 MB [M,D] f32
    _Float16* cnorm = (_Float16*)(ws + 58785792);            //  4 MB [M,D] f16
    _Float16* cact  = (_Float16*)(ws + 62980096);            //  8 MB [M,2D] f16
    float*    cout_ = (float*)   (ws + 71368704);            //  8 MB [M,D] f32

    // 1) positional encoding + shared LN stats + router LN
    prep_kernel<<<M, 256, 0, stream>>>(x, r_ln_g, r_ln_b, h, xhat, rnorm);

    // 2) router FFN1: ract = GELU(rnorm @ r_w1 + r_b1)     [2048,1024]x[1024,4096]
    gemm_wmma<1, false, true><<<dim3(H / 128, M / 128), 256, 0, stream>>>(
        rnorm, D, r_w1, H, r_b1, nullptr, nullptr, nullptr, 0, ract);

    // 3) router head + softmax + top-2 gate
    router_gate_kernel<<<M, 256, 0, stream>>>(ract, r_w2, r_b2, temp, gate);

    // 4) experts (dense, matching reference): per expert two GEMMs.
    for (int e = 0; e < E; ++e) {
        // f1 = GELU((xhat*g_e + b_e) @ e_w1[e] + e_b1[e])  [2048,1024]x[1024,4096]
        // (affine fold-in -> manual A staging path)
        gemm_wmma<1, false, false><<<dim3(HE / 128, M / 128), 256, 0, stream>>>(
            xhat, D, e_w1 + (size_t)e * D * HE, HE, e_b1 + (size_t)e * HE,
            e_ln_g + (size_t)e * D, e_ln_b + (size_t)e * D, nullptr, 0, f1);
        // comb (+)= gate[:,e] * (f1 @ e_w2[e] + e_b2[e])   [2048,4096]x[4096,1024]
        if (e == 0)
            gemm_wmma<3, true, true><<<dim3(D / 128, M / 128), 256, 0, stream>>>(
                f1, HE, e_w2 + (size_t)e * HE * D, D, e_b2 + (size_t)e * D,
                nullptr, nullptr, gate + e, E, comb);
        else
            gemm_wmma<3, false, true><<<dim3(D / 128, M / 128), 256, 0, stream>>>(
                f1, HE, e_w2 + (size_t)e * HE * D, D, e_b2 + (size_t)e * D,
                nullptr, nullptr, gate + e, E, comb);
    }

    // 5) combiner MLP
    ln_f16_kernel<<<M, 256, 0, stream>>>(comb, c_ln_g, c_ln_b, cnorm);
    gemm_wmma<1, false, true><<<dim3(2 * D / 128, M / 128), 256, 0, stream>>>(
        cnorm, D, c_w1, 2 * D, c_b1, nullptr, nullptr, nullptr, 0, cact);
    gemm_wmma<2, false, true><<<dim3(D / 128, M / 128), 256, 0, stream>>>(
        cact, 2 * D, c_w2, D, c_b2, nullptr, nullptr, nullptr, 0, cout_);

    // 6) residual + output LayerNorm
    final_kernel<<<M, 256, 0, stream>>>(h, cout_, o_ln_g, o_ln_b, (float*)d_out);
}
